// RevInAutoencoderLSTM_65678639890635
// MI455X (gfx1250) — matrix operand order repaired and checked by reading, not compiled, not run on hardware
//
#include <hip/hip_runtime.h>
#include <cmath>
#include <cstdint>

// ---------------------------------------------------------------------------
// RevIn autoencoder LSTM for MI455X (gfx1250), wave32, WMMA bf16 path.
//   B=64, seq=335 steps, E=100, H=128, FPS=720, gates 4H=512.
//   4 persistent workgroups, each owns a 16-row batch slice (one WMMA M tile),
//   runs the full recurrence with v_wmma_f32_16x16x32_bf16, activations in
//   LDS, weights streamed from L2 as pre-packed WMMA B-fragments.
//   Paired N-tiles per wave: shared A fragment, two independent accumulators.
// ---------------------------------------------------------------------------

typedef __attribute__((ext_vector_type(16))) __bf16 v16bf;
typedef __attribute__((ext_vector_type(8)))  float  v8f;

struct alignas(16) U4 { unsigned int x, y, z, w; };
union Frag32B { U4 q[2]; v16bf v; };

__device__ __forceinline__ unsigned short f2bf(float f) {
  unsigned int x = __float_as_uint(f);
  unsigned int r = (x + 0x7FFFu + ((x >> 16) & 1u)) >> 16;   // RNE
  return (unsigned short)r;
}

__device__ __forceinline__ float sigmoidf_(float x) {
  return 1.0f / (1.0f + expf(-x));
}

__device__ __forceinline__ v8f vzero8() {
  v8f z = {0.f, 0.f, 0.f, 0.f, 0.f, 0.f, 0.f, 0.f};
  return z;
}

__device__ __forceinline__ v8f wmma_bf16(const Frag32B& a, const Frag32B& b, v8f c) {
  return __builtin_amdgcn_wmma_f32_16x16x32_bf16(
      false, a.v, false, b.v, (short)0, c, false, false);
}

// Single-tile accumulate: C += actLDS(16 x 32*kTiles bf16, stride kStride) @ B.
// B fragments packed so each lane reads 32 contiguous bytes: frag*1024 + lane*32.
__device__ __forceinline__ v8f gemm_acc(v8f acc,
                                        const unsigned short* actLds, int kStride,
                                        const unsigned short* __restrict__ wpack,
                                        int fragBase, int kTiles, int lane) {
  const int m  = lane & 15;
  const int hi = lane >> 4;
  const unsigned short* arow = actLds + m * kStride;
  for (int kt = 0; kt < kTiles; ++kt) {
    Frag32B a, b;
    const unsigned short* ap = arow + kt * 32;
    a.q[0] = *reinterpret_cast<const U4*>(ap + 8 * hi);        // K = 8*hi .. +7
    a.q[1] = *reinterpret_cast<const U4*>(ap + 16 + 8 * hi);   // K = 16+8*hi .. +7
    const unsigned short* wp = wpack + (size_t)(fragBase + kt) * 512 + lane * 16;
    b.q[0] = *reinterpret_cast<const U4*>(wp);
    b.q[1] = *reinterpret_cast<const U4*>(wp + 8);
    if (kt + 1 < kTiles) __builtin_prefetch(wp + 512, 0, 3);   // near-scope prefetch
    acc = wmma_bf16(a, b, acc);
  }
  return acc;
}

// Dual-tile accumulate: two N-tiles sharing one A fragment (half the LDS
// traffic, two independent WMMA accumulation chains for ILP).
__device__ __forceinline__ void gemm_acc2(v8f& acc0, v8f& acc1,
                                          const unsigned short* actLds, int kStride,
                                          const unsigned short* __restrict__ wpack,
                                          int fragBase0, int fragBase1,
                                          int kTiles, int lane) {
  const int m  = lane & 15;
  const int hi = lane >> 4;
  const unsigned short* arow = actLds + m * kStride;
  for (int kt = 0; kt < kTiles; ++kt) {
    Frag32B a, b0, b1;
    const unsigned short* ap = arow + kt * 32;
    a.q[0] = *reinterpret_cast<const U4*>(ap + 8 * hi);
    a.q[1] = *reinterpret_cast<const U4*>(ap + 16 + 8 * hi);
    const unsigned short* wp0 = wpack + (size_t)(fragBase0 + kt) * 512 + lane * 16;
    const unsigned short* wp1 = wpack + (size_t)(fragBase1 + kt) * 512 + lane * 16;
    b0.q[0] = *reinterpret_cast<const U4*>(wp0);
    b0.q[1] = *reinterpret_cast<const U4*>(wp0 + 8);
    b1.q[0] = *reinterpret_cast<const U4*>(wp1);
    b1.q[1] = *reinterpret_cast<const U4*>(wp1 + 8);
    if (kt + 1 < kTiles) {
      __builtin_prefetch(wp0 + 512, 0, 3);
      __builtin_prefetch(wp1 + 512, 0, 3);
    }
    acc0 = wmma_bf16(a, b0, acc0);
    acc1 = wmma_bf16(a, b1, acc1);
  }
}

// ---------------------------------------------------------------------------
// Weight packing: W is (N,K) f32 row-major. Produce bf16 WMMA B-fragments:
//   frag (nt,kt) -> idx = (nt*kT + kt)*512 + lane*16 + e
//   value = W[nt*16 + (lane&15)][kt*32 + 8*(lane>>4) + (e&7) + 16*(e>>3)]  (0 if OOB)
// ---------------------------------------------------------------------------
__global__ void pack_weight_bf16(const float* __restrict__ src,
                                 unsigned short* __restrict__ dst,
                                 int N, int K, int nT, int kT) {
  const int total = nT * kT * 512;
  for (int idx = blockIdx.x * blockDim.x + threadIdx.x; idx < total;
       idx += gridDim.x * blockDim.x) {
    int frag = idx >> 9;
    int lane = (idx >> 4) & 31;
    int e    = idx & 15;
    int nt   = frag / kT;
    int kt   = frag - nt * kT;
    int n = nt * 16 + (lane & 15);
    int k = kt * 32 + 8 * (lane >> 4) + (e & 7) + 16 * (e >> 3);
    float v = (n < N && k < K) ? src[(size_t)n * K + k] : 0.f;
    dst[idx] = f2bf(v);
  }
}

// ---------------------------------------------------------------------------
// Preprocess: segment-reshape + nan_to_num. Writes target_segments (x_seg[:,1:])
// straight into d_out and x_seg[:,0,:] into workspace (only input the net uses).
// ---------------------------------------------------------------------------
__global__ void preprocess_targets(const float* __restrict__ data,
                                   float* __restrict__ targets,
                                   float* __restrict__ seg0) {
  const int TOT = 64 * 336 * 720;
  for (int idx = blockIdx.x * blockDim.x + threadIdx.x; idx < TOT;
       idx += gridDim.x * blockDim.x) {
    int b   = idx / (336 * 720);
    int rem = idx - b * (336 * 720);
    int s   = rem / 720;
    int fp  = rem - s * 720;
    int f   = fp / 30;
    int mm  = fp - f * 30;
    int day = s / 48;
    int seg = s - day * 48;
    int minute = seg * 30 + mm;
    float v = data[(((size_t)b * 7 + day) * 24 + f) * 1440 + minute];
    if (isnan(v)) v = 0.f;
    else if (isinf(v)) v = (v > 0.f) ? 3.402823466e+38f : -3.402823466e+38f;
    if (s == 0) seg0[(size_t)b * 720 + fp] = v;
    else        targets[((size_t)b * 335 + (s - 1)) * 720 + fp] = v;
  }
}

// ---------------------------------------------------------------------------
// Persistent recurrent kernel: 4 WGs x 512 threads (16 waves), M=16 each.
// ---------------------------------------------------------------------------
__global__ void __launch_bounds__(512)
lstm_persistent(const float* __restrict__ seg0,
                const unsigned short* __restrict__ pWih0,
                const unsigned short* __restrict__ pWhh0,
                const unsigned short* __restrict__ pWih1,
                const unsigned short* __restrict__ pWhh1,
                const unsigned short* __restrict__ pDec,
                const unsigned short* __restrict__ pEnc,
                const float* __restrict__ bih0, const float* __restrict__ bhh0,
                const float* __restrict__ bih1, const float* __restrict__ bhh1,
                const float* __restrict__ dec_b, const float* __restrict__ enc_b,
                const float* __restrict__ out_W, const float* __restrict__ out_b,
                float* __restrict__ seq_out, float* __restrict__ label_out) {
  extern __shared__ char smem[];
  unsigned short* s_encx = (unsigned short*)smem;          // 16x128 bf16 (K pad of E=100)
  unsigned short* s_h0b  = s_encx + 16 * 128;              // 16x128 bf16
  unsigned short* s_h1b  = s_h0b  + 16 * 128;              // 16x128 bf16
  unsigned short* s_dec  = s_h1b  + 16 * 128;              // 16x736 bf16 (720 pad)
  float* s_gates = (float*)(s_dec + 16 * 736);             // 16x512 f32
  float* s_c0    = s_gates + 16 * 512;                     // 16x128 f32
  float* s_c1    = s_c0 + 16 * 128;
  float* s_h1f   = s_c1 + 16 * 128;                        // final h1 (f32)

  const int tid   = threadIdx.x;
  const int lane  = tid & 31;
  const int wave  = tid >> 5;
  const int hi    = lane >> 4;
  const int mBase = blockIdx.x * 16;

  for (int i = tid; i < 16 * 128; i += 512) {
    s_c0[i] = 0.f; s_c1[i] = 0.f; s_h1f[i] = 0.f;
    s_h0b[i] = 0;  s_h1b[i] = 0;  s_encx[i] = 0;
  }
  for (int i = tid; i < 16 * 736; i += 512) s_dec[i] = 0;
  __syncthreads();
  for (int i = tid; i < 16 * 720; i += 512) {
    int m = i / 720, c = i - m * 720;
    s_dec[m * 736 + c] = f2bf(seg0[(size_t)(mBase + m) * 720 + c]);
  }
  __syncthreads();

  // stage: encode s_dec (16x720 bf16) -> s_encx (16x100, padded to 128)
  auto stage_encode = [&]() {
    for (int nt = wave; nt < 7; nt += 16) {
      v8f acc = vzero8();
      acc = gemm_acc(acc, s_dec, 736, pEnc, nt * 23, 23, lane);
      int n = nt * 16 + (lane & 15);
      if (n < 100) {
        float bias = enc_b[n];
        for (int r = 0; r < 8; ++r)
          s_encx[(r + 8 * hi) * 128 + n] = f2bf(acc[r] + bias);
      }
    }
  };

  // stage: gates = actX@WihP + h_prev@WhhP + (bih+bhh), 32 N-tiles as 16 pairs
  auto stage_gates = [&](const unsigned short* actX,
                         const unsigned short* h_prev,
                         const unsigned short* WihP, const unsigned short* WhhP,
                         const float* bih, const float* bhh) {
    const int nt0 = wave;        // tiles nt0 and nt0+16 share A fragments
    const int nt1 = wave + 16;
    v8f acc0 = vzero8(), acc1 = vzero8();
    gemm_acc2(acc0, acc1, actX,   128, WihP, nt0 * 4, nt1 * 4, 4, lane);
    gemm_acc2(acc0, acc1, h_prev, 128, WhhP, nt0 * 4, nt1 * 4, 4, lane);
    int n0 = nt0 * 16 + (lane & 15);
    int n1 = nt1 * 16 + (lane & 15);
    float b0 = bih[n0] + bhh[n0];
    float b1 = bih[n1] + bhh[n1];
    for (int r = 0; r < 8; ++r) {
      int row = (r + 8 * hi) * 512;
      s_gates[row + n0] = acc0[r] + b0;
      s_gates[row + n1] = acc1[r] + b1;
    }
  };

  // stage: LSTM elementwise cell
  auto stage_cell = [&](float* s_c, unsigned short* s_hb, float* s_hf) {
    for (int i = tid; i < 2048; i += 512) {
      int m = i >> 7, h = i & 127;
      const float* g = s_gates + m * 512;
      float ig = sigmoidf_(g[h]);
      float fg = sigmoidf_(g[128 + h]);
      float gg = tanhf(g[256 + h]);
      float og = sigmoidf_(g[384 + h]);
      float c = fg * s_c[i] + ig * gg;
      s_c[i] = c;
      float hv = og * tanhf(c);
      if (s_hf) s_hf[i] = hv;
      s_hb[i] = f2bf(hv);
    }
  };

  stage_encode();          // enc0
  __syncthreads();

  for (int t = 0; t < 335; ++t) {
    // ---- layer 0 ----
    stage_gates(s_encx, s_h0b, pWih0, pWhh0, bih0, bhh0);
    __syncthreads();
    stage_cell(s_c0, s_h0b, nullptr);
    __syncthreads();
    // ---- layer 1 ----
    stage_gates(s_h0b, s_h1b, pWih1, pWhh1, bih1, bhh1);
    __syncthreads();
    stage_cell(s_c1, s_h1b, s_h1f);
    __syncthreads();
    // ---- decode: h1@dec_W^T + dec_b -> seq_out[b][t][:] and s_dec ----
    {
      // paired tiles 0..31
      const int nt0 = wave, nt1 = wave + 16;
      v8f acc0 = vzero8(), acc1 = vzero8();
      gemm_acc2(acc0, acc1, s_h1b, 128, pDec, nt0 * 4, nt1 * 4, 4, lane);
      int n0 = nt0 * 16 + (lane & 15);
      int n1 = nt1 * 16 + (lane & 15);
      float b0 = dec_b[n0], b1 = dec_b[n1];
      for (int r = 0; r < 8; ++r) {
        int m = r + 8 * hi;
        size_t orow = ((size_t)(mBase + m) * 335 + t) * 720;
        float v0 = acc0[r] + b0;
        float v1 = acc1[r] + b1;
        seq_out[orow + n0] = v0;
        seq_out[orow + n1] = v1;
        s_dec[m * 736 + n0] = f2bf(v0);
        s_dec[m * 736 + n1] = f2bf(v1);
      }
      // remainder tiles 32..44 (waves 0..12)
      if (wave < 13) {
        int nt = 32 + wave;
        v8f acc = vzero8();
        acc = gemm_acc(acc, s_h1b, 128, pDec, nt * 4, 4, lane);
        int n = nt * 16 + (lane & 15);
        float bias = dec_b[n];
        for (int r = 0; r < 8; ++r) {
          int m = r + 8 * hi;
          float v = acc[r] + bias;
          seq_out[((size_t)(mBase + m) * 335 + t) * 720 + n] = v;
          s_dec[m * 736 + n] = f2bf(v);
        }
      }
    }
    __syncthreads();
    // ---- re-encode for next step ----
    stage_encode();
    __syncthreads();
  }

  // label_pred = h1_final @ out_W^T + out_b
  if (tid < 16) {
    float s = out_b[0];
    const float* hrow = s_h1f + tid * 128;
    for (int j = 0; j < 128; ++j) s += hrow[j] * out_W[j];
    label_out[mBase + tid] = s;
  }
}

// ---------------------------------------------------------------------------
extern "C" void kernel_launch(void* const* d_in, const int* in_sizes, int n_in,
                              void* d_out, int out_size, void* d_ws, size_t ws_size,
                              hipStream_t stream) {
  (void)in_sizes; (void)n_in; (void)out_size; (void)ws_size;
  const float* data  = (const float*)d_in[0];
  const float* enc_W = (const float*)d_in[1];   // (100,720)
  const float* enc_b = (const float*)d_in[2];
  const float* Wih0  = (const float*)d_in[3];   // (512,100)
  const float* Whh0  = (const float*)d_in[4];   // (512,128)
  const float* bih0  = (const float*)d_in[5];
  const float* bhh0  = (const float*)d_in[6];
  const float* Wih1  = (const float*)d_in[7];   // (512,128)
  const float* Whh1  = (const float*)d_in[8];   // (512,128)
  const float* bih1  = (const float*)d_in[9];
  const float* bhh1  = (const float*)d_in[10];
  const float* dec_W = (const float*)d_in[11];  // (720,128)
  const float* dec_b = (const float*)d_in[12];
  const float* out_W = (const float*)d_in[13];  // (1,128)
  const float* out_b = (const float*)d_in[14];

  // workspace carve (bf16 packed weights + cleaned first segment)
  unsigned short* pWih0 = (unsigned short*)d_ws;
  unsigned short* pWhh0 = pWih0 + 32 * 4 * 512;     // 65536 elems each
  unsigned short* pWih1 = pWhh0 + 32 * 4 * 512;
  unsigned short* pWhh1 = pWih1 + 32 * 4 * 512;
  unsigned short* pDec  = pWhh1 + 32 * 4 * 512;     // 45*4*512 = 92160
  unsigned short* pEnc  = pDec  + 45 * 4 * 512;     // 7*23*512 = 82432
  float*          seg0  = (float*)(pEnc + 7 * 23 * 512);

  float* seq_out = (float*)d_out;                       // (64,335,720)
  float* targets = seq_out + (size_t)64 * 335 * 720;    // (64,335,720)
  float* label   = targets + (size_t)64 * 335 * 720;    // (64,)

  auto blocks = [](int n) { return (n + 255) / 256; };
  pack_weight_bf16<<<blocks(32 * 4 * 512), 256, 0, stream>>>(Wih0, pWih0, 512, 100, 32, 4);
  pack_weight_bf16<<<blocks(32 * 4 * 512), 256, 0, stream>>>(Whh0, pWhh0, 512, 128, 32, 4);
  pack_weight_bf16<<<blocks(32 * 4 * 512), 256, 0, stream>>>(Wih1, pWih1, 512, 128, 32, 4);
  pack_weight_bf16<<<blocks(32 * 4 * 512), 256, 0, stream>>>(Whh1, pWhh1, 512, 128, 32, 4);
  pack_weight_bf16<<<blocks(45 * 4 * 512), 256, 0, stream>>>(dec_W, pDec, 720, 128, 45, 4);
  pack_weight_bf16<<<blocks(7 * 23 * 512), 256, 0, stream>>>(enc_W, pEnc, 100, 720, 7, 23);

  preprocess_targets<<<2048, 256, 0, stream>>>(data, targets, seg0);

  const size_t smemBytes = (16 * 128 * 3 + 16 * 736) * sizeof(unsigned short)
                         + (16 * 512 + 16 * 128 * 3) * sizeof(float);  // 93184 B
  lstm_persistent<<<4, 512, smemBytes, stream>>>(
      seg0, pWih0, pWhh0, pWih1, pWhh1, pDec, pEnc,
      bih0, bhh0, bih1, bhh1, dec_b, enc_b, out_W, out_b,
      seq_out, label);
}
